// GINNet_29678224015468
// MI455X (gfx1250) — matrix operand order repaired and checked by reading, not compiled, not run on hardware
//
#include <hip/hip_runtime.h>
#include <math.h>

// ---------------------------------------------------------------------------
// GIN net for MI455X (gfx1250, wave32, WMMA).
//   y1   = x @ W1a                          (WMMA f16 16x16x32, K=128)
//   agg  = segment_sum(y1[src] -> dst)      (L2-resident f32 atomics)
//   h    = relu( relu(y1+agg+b1a) @ W1b + b1b )   (WMMA, + BN stats)
//   h    = BN(h)
//   agg2 = segment_sum(h[src] -> dst)
//   out  = relu((h+agg2) @ W2a + b2a) @ W2b + b2b (WMMA x2, LDS transpose)
// Weights are pre-swizzled in LDS into per-lane WMMA-fragment order so each
// B operand is a single contiguous 32B v16h load (2x ds_load_b128).
// ---------------------------------------------------------------------------

typedef __attribute__((ext_vector_type(16))) _Float16 v16h;
typedef __attribute__((ext_vector_type(8)))  _Float16 v8h;
typedef __attribute__((ext_vector_type(8)))  float    v8f;

__device__ __forceinline__ float relu(float v) { return v > 0.0f ? v : 0.0f; }

#define WMMA_F16(A, B, C) \
  __builtin_amdgcn_wmma_f32_16x16x32_f16(false, (A), false, (B), (short)0, (C), false, false)

// Swizzle a [Kdim x 32] row-major f32 weight into LDS fragment order:
//   dst[((c*2+nt)*32 + lane)*16 + e] = W[(c*32 + 8*(lane>>4) + (e&7) + 16*(e>>3))*32 + (lane&15) + 16*nt]
__device__ __forceinline__ void swizzleW(const float* __restrict__ W,
                                         _Float16* __restrict__ sW,
                                         int nElem /* = Kdim*32 */) {
  for (int j = threadIdx.x; j < nElem; j += 256) {
    const int e  = j & 15;
    const int l  = (j >> 4) & 31;
    const int nt = (j >> 9) & 1;
    const int c  = j >> 10;
    const int K  = c * 32 + 8 * (l >> 4) + (e & 7) + 16 * (e >> 3);
    const int col = (l & 15) + 16 * nt;
    sW[j] = (_Float16)W[K * 32 + col];
  }
}

// ------------------------- y1 = x @ W1a  (N x 128 x 32) ---------------------
__global__ __launch_bounds__(256) void k_proj1(const float* __restrict__ x,
                                               const float* __restrict__ W1a,
                                               float* __restrict__ y1,
                                               int nTiles, int Nn) {
  __shared__ _Float16 sW[128 * 32];      // fragment order, 8 fragments
  swizzleW(W1a, sW, 128 * 32);
  __syncthreads();

  const int wave = threadIdx.x >> 5, lane = threadIdx.x & 31;
  const int tile = blockIdx.x * 8 + wave;
  if (tile >= nTiles) return;

  const int rowl  = lane & 15;
  const int khalf = lane >> 4;
  int row = tile * 16 + rowl;
  if (row >= Nn) row = Nn - 1;          // clamp (N is a multiple of 16 here)
  const float* __restrict__ xr = x + (size_t)row * 128;

  v8f acc0 = {0.f,0.f,0.f,0.f,0.f,0.f,0.f,0.f};
  v8f acc1 = {0.f,0.f,0.f,0.f,0.f,0.f,0.f,0.f};

#pragma unroll
  for (int c = 0; c < 4; ++c) {
    // A fragment: 16x32 f16 (ISA 16-bit A layout; lane half selects K range).
    v16h a;
    const int K0 = c * 32 + 8 * khalf;
    const int K1 = c * 32 + 16 + 8 * khalf;
#pragma unroll
    for (int i = 0; i < 8; ++i) {
      a[i]     = (_Float16)xr[K0 + i];
      a[8 + i] = (_Float16)xr[K1 + i];
    }
    // B fragments: contiguous 32B per lane (pre-swizzled).
    const v16h b0 = *(const v16h*)(sW + ((c * 2 + 0) * 32 + lane) * 16);
    const v16h b1 = *(const v16h*)(sW + ((c * 2 + 1) * 32 + lane) * 16);
    acc0 = WMMA_F16(a, b0, acc0);
    acc1 = WMMA_F16(a, b1, acc1);
  }
#pragma unroll
  for (int r = 0; r < 8; ++r) {
    const int ro = tile * 16 + r + 8 * khalf;
    if (ro < Nn) {
      y1[(size_t)ro * 32 + rowl]      = acc0[r];
      y1[(size_t)ro * 32 + rowl + 16] = acc1[r];
    }
  }
}

// ----------------- agg[dst] += val[src]  (32 floats / edge) -----------------
__global__ __launch_bounds__(256) void k_agg(const long long* __restrict__ src,
                                             const long long* __restrict__ dst,
                                             const float* __restrict__ val,
                                             float* __restrict__ acc, int E_) {
  const int t = blockIdx.x * blockDim.x + threadIdx.x;
  const int e = t >> 3, p = t & 7;
  if (e >= E_) return;
  __builtin_prefetch(src + e + 4096, 0, 1);   // global_prefetch_b8 on edge list
  __builtin_prefetch(dst + e + 4096, 0, 1);
  const int s = (int)src[e];
  const int d = (int)dst[e];
  const float4 v = *(const float4*)(val + (size_t)s * 32 + p * 4);
  float* a = acc + (size_t)d * 32 + p * 4;
  atomicAdd(a + 0, v.x);
  atomicAdd(a + 1, v.y);
  atomicAdd(a + 2, v.z);
  atomicAdd(a + 3, v.w);
}

// --- h_pre = relu( relu(y1+agg+b1a) @ W1b + b1b ); accumulate BN stats ------
__global__ __launch_bounds__(256) void k_mlp1(const float* __restrict__ y1,
                                              const float* __restrict__ agg,
                                              const float* __restrict__ b1a,
                                              const float* __restrict__ W1b,
                                              const float* __restrict__ b1b,
                                              float* __restrict__ hout,
                                              float* __restrict__ stats,
                                              int nTiles, int Nn) {
  __shared__ _Float16 sW[32 * 32];       // fragment order, 2 fragments
  swizzleW(W1b, sW, 32 * 32);
  __syncthreads();

  const int wave = threadIdx.x >> 5, lane = threadIdx.x & 31;
  const int tile = blockIdx.x * 8 + wave;
  if (tile >= nTiles) return;

  const int rowl = lane & 15, khalf = lane >> 4;
  int row = tile * 16 + rowl;
  if (row >= Nn) row = Nn - 1;
  const float* __restrict__ p0 = y1 + (size_t)row * 32;
  const float* __restrict__ p1 = agg + (size_t)row * 32;

  v16h a;
  const int K0 = 8 * khalf, K1 = 16 + 8 * khalf;
#pragma unroll
  for (int i = 0; i < 8; ++i) {
    a[i]     = (_Float16)relu(p0[K0 + i] + p1[K0 + i] + b1a[K0 + i]);
    a[8 + i] = (_Float16)relu(p0[K1 + i] + p1[K1 + i] + b1a[K1 + i]);
  }
  const v16h b0 = *(const v16h*)(sW + (size_t)lane * 16);
  const v16h b1 = *(const v16h*)(sW + (size_t)(32 + lane) * 16);

  v8f c0 = {0.f,0.f,0.f,0.f,0.f,0.f,0.f,0.f};
  v8f c1 = {0.f,0.f,0.f,0.f,0.f,0.f,0.f,0.f};
  c0 = WMMA_F16(a, b0, c0);
  c1 = WMMA_F16(a, b1, c1);

  const float bias0 = b1b[rowl], bias1 = b1b[rowl + 16];
  float s0 = 0.f, q0 = 0.f, s1 = 0.f, q1 = 0.f;
#pragma unroll
  for (int r = 0; r < 8; ++r) {
    const float v0 = relu(c0[r] + bias0);
    const float v1 = relu(c1[r] + bias1);
    const int ro = tile * 16 + r + 8 * khalf;
    if (ro < Nn) {
      hout[(size_t)ro * 32 + rowl]      = v0;
      hout[(size_t)ro * 32 + rowl + 16] = v1;
    }
    s0 += v0; q0 += v0 * v0;
    s1 += v1; q1 += v1 * v1;
  }
  atomicAdd(&stats[rowl],           s0);
  atomicAdd(&stats[32 + rowl],      q0);
  atomicAdd(&stats[rowl + 16],      s1);
  atomicAdd(&stats[32 + rowl + 16], q1);
}

// --------------- fold BN: stats -> per-channel scale / shift ----------------
__global__ void k_bnfin(const float* __restrict__ gamma,
                        const float* __restrict__ beta,
                        float* __restrict__ stats, float invN) {
  const int j = threadIdx.x;  // 32 threads
  const float mean = stats[j] * invN;
  const float var  = stats[32 + j] * invN - mean * mean;
  const float sc   = gamma[j] * rsqrtf(var + 1e-5f);
  stats[64 + j] = sc;
  stats[96 + j] = beta[j] - mean * sc;
}

// ------------------------- h = h*scale + shift (in place) -------------------
__global__ __launch_bounds__(256) void k_bnapply(float* __restrict__ h,
                                                 const float* __restrict__ stats,
                                                 int n8) {
  const int t = blockIdx.x * blockDim.x + threadIdx.x;
  if (t >= n8) return;
  const int p = t & 7;
  const float4 sc = *(const float4*)(stats + 64 + 4 * p);
  const float4 sh = *(const float4*)(stats + 96 + 4 * p);
  float4 v = ((float4*)h)[t];
  v.x = v.x * sc.x + sh.x;
  v.y = v.y * sc.y + sh.y;
  v.z = v.z * sc.z + sh.z;
  v.w = v.w * sc.w + sh.w;
  ((float4*)h)[t] = v;
}

// ---- out = relu((h+agg2) @ W2a + b2a) @ W2b + b2b  (two chained WMMAs) -----
__global__ __launch_bounds__(256) void k_mlp2(const float* __restrict__ h,
                                              const float* __restrict__ agg,
                                              const float* __restrict__ W2a,
                                              const float* __restrict__ b2a,
                                              const float* __restrict__ W2b,
                                              const float* __restrict__ b2b,
                                              float* __restrict__ out,
                                              int nTiles, int Nn) {
  __shared__ _Float16 sWa[32 * 32];      // fragment order, 2 fragments
  __shared__ _Float16 sWb[32 * 16];      // fragment order, 1 fragment (10->16 pad)
  __shared__ _Float16 sU[8][16 * 40];    // per-wave u tile (C->A transpose)
  swizzleW(W2a, sWa, 32 * 32);
  for (int j = threadIdx.x; j < 32 * 16; j += 256) {
    const int e = j & 15, l = j >> 4;
    const int K = 8 * (l >> 4) + (e & 7) + 16 * (e >> 3);
    const int col = l & 15;
    sWb[j] = (col < 10) ? (_Float16)W2b[K * 10 + col] : (_Float16)0.0f;
  }
  __syncthreads();

  const int wave = threadIdx.x >> 5, lane = threadIdx.x & 31;
  const int tile = blockIdx.x * 8 + wave;
  if (tile >= nTiles) return;

  const int rowl = lane & 15, khalf = lane >> 4;
  int row = tile * 16 + rowl;
  if (row >= Nn) row = Nn - 1;
  const float* __restrict__ p0 = h + (size_t)row * 32;
  const float* __restrict__ p1 = agg + (size_t)row * 32;

  // GEMM 1: (h + agg2) @ W2a
  v16h a;
  const int K0 = 8 * khalf, K1 = 16 + 8 * khalf;
#pragma unroll
  for (int i = 0; i < 8; ++i) {
    a[i]     = (_Float16)(p0[K0 + i] + p1[K0 + i]);
    a[8 + i] = (_Float16)(p0[K1 + i] + p1[K1 + i]);
  }
  const v16h b0 = *(const v16h*)(sWa + (size_t)lane * 16);
  const v16h b1 = *(const v16h*)(sWa + (size_t)(32 + lane) * 16);
  v8f c0 = {0.f,0.f,0.f,0.f,0.f,0.f,0.f,0.f};
  v8f c1 = {0.f,0.f,0.f,0.f,0.f,0.f,0.f,0.f};
  c0 = WMMA_F16(a, b0, c0);
  c1 = WMMA_F16(a, b1, c1);

  // relu(+b2a), bounce through LDS to transpose C layout -> A layout.
  // Row stride 40 halves (80B) keeps every v8h sub-load 16B aligned.
  const float bias0 = b2a[rowl], bias1 = b2a[rowl + 16];
  _Float16* us = sU[wave];
#pragma unroll
  for (int r = 0; r < 8; ++r) {
    const int rr = r + 8 * khalf;
    us[rr * 40 + rowl]      = (_Float16)relu(c0[r] + bias0);
    us[rr * 40 + rowl + 16] = (_Float16)relu(c1[r] + bias1);
  }
  asm volatile("s_wait_dscnt 0" ::: "memory");   // wave-local LDS RAW fence

  // GEMM 2: u @ W2b  (cols >= 10 are zero-padded)
  const v8h lo = *(const v8h*)(us + rowl * 40 + K0);
  const v8h hi = *(const v8h*)(us + rowl * 40 + K1);
  const v16h a2 = __builtin_shufflevector(lo, hi,
      0, 1, 2, 3, 4, 5, 6, 7, 8, 9, 10, 11, 12, 13, 14, 15);
  const v16h b2 = *(const v16h*)(sWb + (size_t)lane * 16);
  v8f c2 = {0.f,0.f,0.f,0.f,0.f,0.f,0.f,0.f};
  c2 = WMMA_F16(a2, b2, c2);

  if (rowl < 10) {
    const float bias = b2b[rowl];
#pragma unroll
    for (int r = 0; r < 8; ++r) {
      const int ro = tile * 16 + r + 8 * khalf;
      if (ro < Nn) out[(size_t)ro * 10 + rowl] = c2[r] + bias;
    }
  }
}

// ---------------------------------------------------------------------------
extern "C" void kernel_launch(void* const* d_in, const int* in_sizes, int n_in,
                              void* d_out, int out_size, void* d_ws, size_t ws_size,
                              hipStream_t stream) {
  const float*     x    = (const float*)d_in[0];
  const long long* ei   = (const long long*)d_in[1];
  // d_in[2] = batch (unused: single graph, per-node output)
  const float* W1a  = (const float*)d_in[3];
  const float* b1a  = (const float*)d_in[4];
  const float* W1b  = (const float*)d_in[5];
  const float* b1b  = (const float*)d_in[6];
  const float* gamma= (const float*)d_in[7];
  const float* beta = (const float*)d_in[8];
  const float* W2a  = (const float*)d_in[9];
  const float* b2a  = (const float*)d_in[10];
  const float* W2b  = (const float*)d_in[11];
  const float* b2b  = (const float*)d_in[12];
  float* out = (float*)d_out;

  const int N = in_sizes[0] / 128;
  const int E = in_sizes[1] / 2;
  const long long* src = ei;
  const long long* dst = ei + E;

  float* y1    = (float*)d_ws;                 // [N,32]
  float* agg   = y1  + (size_t)N * 32;         // [N,32] (reused for both aggs)
  float* hbuf  = agg + (size_t)N * 32;         // [N,32]
  float* stats = hbuf + (size_t)N * 32;        // [128]: sum|sumsq|scale|shift

  const int nTiles    = (N + 15) / 16;
  const int mlpBlocks = (nTiles + 7) / 8;
  const int aggGrid   = (E * 8 + 255) / 256;

  hipMemsetAsync(agg,   0, (size_t)N * 32 * sizeof(float), stream);
  hipMemsetAsync(stats, 0, 128 * sizeof(float), stream);

  // conv1 projected to DIM early (linearity of GIN sum): 4x less agg traffic
  k_proj1<<<mlpBlocks, 256, 0, stream>>>(x, W1a, y1, nTiles, N);
  k_agg  <<<aggGrid,   256, 0, stream>>>(src, dst, y1, agg, E);
  k_mlp1 <<<mlpBlocks, 256, 0, stream>>>(y1, agg, b1a, W1b, b1b, hbuf, stats, nTiles, N);

  // batchnorm (training-mode batch statistics)
  k_bnfin  <<<1, 32, 0, stream>>>(gamma, beta, stats, 1.0f / (float)N);
  k_bnapply<<<(N * 8 + 255) / 256, 256, 0, stream>>>(hbuf, stats, N * 8);

  // conv2
  hipMemsetAsync(agg, 0, (size_t)N * 32 * sizeof(float), stream);
  k_agg <<<aggGrid,   256, 0, stream>>>(src, dst, hbuf, agg, E);
  k_mlp2<<<mlpBlocks, 256, 0, stream>>>(hbuf, agg, W2a, b2a, W2b, b2b, out, nTiles, N);
}
